// KromHCHeadMixer_4724464025795
// MI455X (gfx1250) — matrix operand order repaired and checked by reading, not compiled, not run on hardware
//
#include <hip/hip_runtime.h>

// KromHCHeadMixer for MI455X (gfx1250, wave32).
// out = (U0 (x) U1 (x) U2) @ x ; also emits H = U0 (x) U1 (x) U2.
// First-layer GEMM (context @ W1) runs on v_wmma_f32_16x16x32_bf16.

static constexpr int BSZ    = 131072;
static constexpr int NH     = 8;
static constexpr int HD     = 64;
static constexpr int DC     = 64;   // context dim == HEAD_DIM
static constexpr int KMLP   = 3;
static constexpr int ROWS   = 64;   // batch rows per block
static constexpr int THREADS= 128;  // 4 waves (wave32)
static constexpr int WAVES  = 4;

typedef __bf16  bf16_t;
typedef bf16_t  v16bf __attribute__((ext_vector_type(16)));
typedef float   v8f   __attribute__((ext_vector_type(8)));

__device__ __forceinline__ bf16_t f2bf(float f) {
  // round-to-nearest-even fp32 -> bf16
  unsigned u = __builtin_bit_cast(unsigned, f);
  unsigned r = (u + 0x7FFFu + ((u >> 16) & 1u)) >> 16;
  unsigned short s = (unsigned short)r;
  return __builtin_bit_cast(bf16_t, s);
}

__global__ __launch_bounds__(THREADS)
void krom_mixer_kernel(const float* __restrict__ x,
                       const float* __restrict__ W1,
                       const float* __restrict__ b1,
                       const float* __restrict__ W2,
                       const float* __restrict__ b2,
                       float* __restrict__ out,
                       float* __restrict__ Hout)
{
  __shared__ float s_ctx[ROWS][DC + 1];     // +1 pad: distinct banks per row
  __shared__ float s_h[WAVES][16][33];      // relu(h) staging per wave tile
  __shared__ float s_a[ROWS][KMLP];         // a0 per (row, mlp)

  const int t    = threadIdx.x;
  const int row0 = blockIdx.x * ROWS;

  // ---------- Phase 0: context = mean over heads (coalesced) ----------
  for (int idx = t; idx < ROWS * DC; idx += THREADS) {
    const int b = idx >> 6;
    const int d = idx & 63;
    const float* xb = x + (size_t)(row0 + b) * (NH * HD) + d;
    float s = 0.f;
    #pragma unroll
    for (int c = 0; c < NH; ++c) s += xb[c * HD];
    s_ctx[b][d] = s * 0.125f;
  }
  __syncthreads();

  const int lane = t & 31;
  const int w    = t >> 5;
  const int hsel = lane >> 4;   // half-wave selector (ISA 16-bit frag layout)
  const int l15  = lane & 15;

  // ---------- Phase 1: three tiny MLPs; layer 1 on WMMA ----------
  for (int i = 0; i < KMLP; ++i) {
    // A: context tile, 16 rows x 64 K, split into two 16x32 bf16 fragments.
    // 16-bit A layout: lane<16 -> M=lane, elems 0..7 = K(8h..8h+7),
    // elems 8..15 = K(16+8h..); lane>=16 shifts K by 8 (h = lane>>4).
    const int m = w * 16 + l15;
    v16bf a0, a1;
    #pragma unroll
    for (int j = 0; j < 8; ++j) {
      const int kb = hsel * 8 + j;
      a0[j]     = f2bf(s_ctx[m][kb]);
      a0[j + 8] = f2bf(s_ctx[m][16 + kb]);
      a1[j]     = f2bf(s_ctx[m][32 + kb]);
      a1[j + 8] = f2bf(s_ctx[m][48 + kb]);
    }
    #pragma unroll
    for (int nt = 0; nt < 2; ++nt) {
      // B: W1[i] 32x16 tile per K-chunk.  16-bit B layout:
      // lane -> N = lane&15, elem j<8 -> K = (lane>>4)*8 + j, j>=8 -> K = 16 + (lane>>4)*8 + (j-8).
      const int n = nt * 16 + l15;
      const float* w1p = W1 + (size_t)i * 64 * 32 + n;
      v16bf bf0, bf1;
      #pragma unroll
      for (int j = 0; j < 16; ++j) {
        const int k0 = (j < 8) ? (hsel * 8 + j) : (16 + hsel * 8 + (j - 8));
        bf0[j] = f2bf(w1p[k0 * 32]);          // K-chunk 0: k = 0..31
        bf1[j] = f2bf(w1p[(32 + k0) * 32]);   // K-chunk 1: k = 32..63
      }
      v8f acc = {};
      acc = __builtin_amdgcn_wmma_f32_16x16x32_bf16(false, a0, false, bf0,
                                                    (short)0, acc, false, false);
      acc = __builtin_amdgcn_wmma_f32_16x16x32_bf16(false, a1, false, bf1,
                                                    (short)0, acc, false, false);
      // C/D layout: VGPR r, lane -> (M = r + 8*(lane>>4), N = lane&15)
      const float bias = b1[i * 32 + n];
      #pragma unroll
      for (int r = 0; r < 8; ++r) {
        const int ml = r + 8 * hsel;
        const float hv = acc[r] + bias;
        s_h[w][ml][n] = hv > 0.f ? hv : 0.f;
      }
    }
    __syncthreads();

    // Layer 2 (32 -> 2) + softmax over 2 == sigmoid(l0 - l1).
    if (t < ROWS) {
      const float* hrow = s_h[t >> 4][t & 15];
      const float* w2p  = W2 + (size_t)i * 32 * 2;
      float l0 = b2[i * 2 + 0];
      float l1 = b2[i * 2 + 1];
      #pragma unroll
      for (int n = 0; n < 32; ++n) {
        const float hv = hrow[n];
        l0 += hv * w2p[n * 2 + 0];
        l1 += hv * w2p[n * 2 + 1];
      }
      s_a[t][i] = 1.f / (1.f + __expf(l1 - l0));
    }
    __syncthreads();
  }

  // ---------- Phase 2a: H[r,c] = prod_i U_i[bit_i(r), bit_i(c)] ----------
  for (int idx = t; idx < ROWS * 64; idx += THREADS) {
    const int b = idx >> 6;
    const int e = idx & 63;
    const int r = e >> 3, c = e & 7;
    const float p0 = s_a[b][0], p1 = s_a[b][1], p2 = s_a[b][2];
    const float u0 = (((r >> 2) ^ (c >> 2)) & 1) ? (1.f - p0) : p0;
    const float u1 = (((r >> 1) ^ (c >> 1)) & 1) ? (1.f - p1) : p1;
    const float u2 = ((r ^ c) & 1) ? (1.f - p2) : p2;
    Hout[(size_t)(row0 + b) * 64 + e] = u0 * u1 * u2;
  }

  // ---------- Phase 2b: out = H @ x as 3 Kronecker butterfly stages ----
  for (int idx = t; idx < ROWS * HD; idx += THREADS) {
    const int b = idx >> 6;
    const int d = idx & 63;
    const float* xb = x + (size_t)(row0 + b) * (NH * HD) + d;  // L2-hot re-read
    float v[8];
    #pragma unroll
    for (int c = 0; c < 8; ++c) v[c] = xb[c * HD];
    const float p0 = s_a[b][0], p1 = s_a[b][1], p2 = s_a[b][2];
    #pragma unroll
    for (int p = 0; p < 8; p += 2) {            // U2: stride-1 pairs
      const float va = v[p], vb = v[p + 1];
      v[p]     = p2 * va + (1.f - p2) * vb;
      v[p + 1] = (1.f - p2) * va + p2 * vb;
    }
    #pragma unroll
    for (int g = 0; g < 8; g += 4)              // U1: stride-2 pairs
      #pragma unroll
      for (int p = 0; p < 2; ++p) {
        const float va = v[g + p], vb = v[g + p + 2];
        v[g + p]     = p1 * va + (1.f - p1) * vb;
        v[g + p + 2] = (1.f - p1) * va + p1 * vb;
      }
    #pragma unroll
    for (int p = 0; p < 4; ++p) {               // U0: stride-4 pairs
      const float va = v[p], vb = v[p + 4];
      v[p]     = p0 * va + (1.f - p0) * vb;
      v[p + 4] = (1.f - p0) * va + p0 * vb;
    }
    float* ob = out + (size_t)(row0 + b) * (NH * HD) + d;
    #pragma unroll
    for (int r = 0; r < 8; ++r) ob[r * HD] = v[r];
  }
}

extern "C" void kernel_launch(void* const* d_in, const int* in_sizes, int n_in,
                              void* d_out, int out_size, void* d_ws, size_t ws_size,
                              hipStream_t stream) {
  (void)in_sizes; (void)n_in; (void)out_size; (void)d_ws; (void)ws_size;
  const float* x  = (const float*)d_in[0];
  const float* W1 = (const float*)d_in[1];
  const float* b1 = (const float*)d_in[2];
  const float* W2 = (const float*)d_in[3];
  const float* b2 = (const float*)d_in[4];
  float* out  = (float*)d_out;
  float* Hout = out + (size_t)BSZ * NH * HD;   // tuple return: (out, H) flat

  dim3 grid(BSZ / ROWS), block(THREADS);
  hipLaunchKernelGGL(krom_mixer_kernel, grid, block, 0, stream,
                     x, W1, b1, W2, b2, out, Hout);
}
